// FeatureParallelCLT_87239375716482
// MI455X (gfx1250) — matrix operand order repaired and checked by reading, not compiled, not run on hardware
//
#include <hip/hip_runtime.h>

typedef __attribute__((ext_vector_type(16))) __bf16 bf16x16;
typedef __attribute__((ext_vector_type(8)))  __bf16 bf16x8;
typedef __attribute__((ext_vector_type(2)))  __bf16 bf16x2;
typedef __attribute__((ext_vector_type(8)))  float  floatx8;

namespace {
constexpr int kT = 4096;
constexpr int kL = 8;
constexpr int kD = 768;
constexpr int kF = 2048;

constexpr int BM = 128;   // token tile
constexpr int BN = 128;   // output-column tile
constexpr int BK = 32;    // reduction step (one bf16 WMMA K)
constexpr int LSTR = 40;  // LDS row stride in bf16 elems (32 + 8 pad)
}

// Assemble a v16 bf16 fragment from two contiguous 8-element LDS chunks
// (ds_load_b128 each), per the CDNA5 16-bit A/B VGPR layouts.
__device__ __forceinline__ bf16x16 make_frag(const __bf16* p0, const __bf16* p1) {
    const bf16x8 lo = *(const bf16x8*)p0;
    const bf16x8 up = *(const bf16x8*)p1;
    bf16x16 f;
#pragma unroll
    for (int e = 0; e < 8; ++e) { f[e] = lo[e]; f[e + 8] = up[e]; }
    return f;
}

// ---------------------------------------------------------------------------
// Encoder: h[t0:t0+128, l, f0:f0+128] = x[t, l, :] @ w_e[l]   (K = 768)
// plus JumpReLU epilogue: acts = h * (h > exp(t))
// Double-buffered LDS pipeline: global loads for step k+1 are in flight while
// WMMAs consume step k; one barrier per K-step.
// ---------------------------------------------------------------------------
__global__ void __launch_bounds__(256)
clt_encoder_wmma(const float* __restrict__ x,
                 const float* __restrict__ w_e,
                 const float* __restrict__ tparam,
                 float* __restrict__ h,
                 float* __restrict__ acts)
{
    __shared__ __bf16 As[2][BM * LSTR];
    __shared__ __bf16 Bs[2][BN * LSTR];

    const int tid   = threadIdx.x;
    const int lane  = tid & 31;
    const int wave  = tid >> 5;
    const int r     = lane & 15;   // row/col within 16-wide fragment
    const int hi    = lane >> 4;   // half-wave selector (ISA K-split)
    const int waveM = wave & 1;    // 2 waves along M -> 64 rows each
    const int waveN = wave >> 1;   // 4 waves along N -> 32 cols each

    const int f0 = blockIdx.x * BN;
    const int t0 = blockIdx.y * BM;
    const int l  = blockIdx.z;

    const float* Ag = x   + (size_t)t0 * (kL * kD) + (size_t)l * kD;   // row stride kL*kD
    const float* Bg = w_e + (size_t)l * kD * kF + f0;                  // row stride kF

    floatx8 zacc = {};
    floatx8 acc[4][2];
#pragma unroll
    for (int a = 0; a < 4; ++a)
#pragma unroll
        for (int b = 0; b < 2; ++b) acc[a][b] = zacc;

    // A staging: each thread = one row x 4 consecutive k (float4)
    const int a_row = tid >> 3;          // 0..31
    const int a_col = (tid & 7) << 2;    // 0,4,..,28
    // B staging: each thread = 2 consecutive k rows x 2 consecutive n cols
    const int nb = (tid & 63) << 1;      // 0,2,..,126
    const int kb = (tid >> 6) << 1;      // 0,2,4,6  (+ p*8)

    float4 va[4];
    float2 vblo[4], vbhi[4];

    auto load_stage = [&](int k0) {
#pragma unroll
        for (int p = 0; p < 4; ++p)
            va[p] = *(const float4*)(Ag + (size_t)(a_row + p * 32) * (kL * kD) + (k0 + a_col));
#pragma unroll
        for (int p = 0; p < 4; ++p) {
            const int k = kb + p * 8;
            vblo[p] = *(const float2*)(Bg + (size_t)(k0 + k)     * kF + nb);
            vbhi[p] = *(const float2*)(Bg + (size_t)(k0 + k + 1) * kF + nb);
        }
    };
    auto store_stage = [&](int buf) {
#pragma unroll
        for (int p = 0; p < 4; ++p) {
            __bf16* d = &As[buf][(a_row + p * 32) * LSTR + a_col];
            d[0] = (__bf16)va[p].x; d[1] = (__bf16)va[p].y;
            d[2] = (__bf16)va[p].z; d[3] = (__bf16)va[p].w;
        }
#pragma unroll
        for (int p = 0; p < 4; ++p) {
            const int k = kb + p * 8;
            bf16x2 c0, c1;
            c0[0] = (__bf16)vblo[p].x; c0[1] = (__bf16)vbhi[p].x;   // col nb,   k..k+1
            c1[0] = (__bf16)vblo[p].y; c1[1] = (__bf16)vbhi[p].y;   // col nb+1, k..k+1
            *(bf16x2*)&Bs[buf][(nb + 0) * LSTR + k] = c0;
            *(bf16x2*)&Bs[buf][(nb + 1) * LSTR + k] = c1;
        }
    };

    // prologue: stage K-step 0
    load_stage(0);
    store_stage(0);
    __syncthreads();

    constexpr int NK = kD / BK;   // 24
    for (int kk = 0; kk < NK; ++kk) {
        const int cur = kk & 1;
        if (kk + 1 < NK) load_stage((kk + 1) * BK);          // next tile in flight
        if (kk + 2 < NK) {                                    // warm L2 for k+2
            __builtin_prefetch(Ag + (size_t)a_row * (kL * kD) + ((kk + 2) * BK + a_col), 0, 1);
            __builtin_prefetch(Bg + (size_t)((kk + 2) * BK + kb) * kF + nb, 0, 1);
        }

        const __bf16* Ab = As[cur];
        const __bf16* Bb = Bs[cur];
        bf16x16 af[4];
#pragma unroll
        for (int tm = 0; tm < 4; ++tm) {
            const int m = waveM * 64 + tm * 16 + r;
            af[tm] = make_frag(&Ab[m * LSTR + hi * 8], &Ab[m * LSTR + 16 + hi * 8]);
        }
        bf16x16 bfr[2];
#pragma unroll
        for (int tn = 0; tn < 2; ++tn) {
            const int c = waveN * 32 + tn * 16 + r;
            bfr[tn] = make_frag(&Bb[c * LSTR + hi * 8], &Bb[c * LSTR + 16 + hi * 8]);
        }
#pragma unroll
        for (int tm = 0; tm < 4; ++tm)
#pragma unroll
            for (int tn = 0; tn < 2; ++tn)
                acc[tm][tn] = __builtin_amdgcn_wmma_f32_16x16x32_bf16(
                    false, af[tm], false, bfr[tn], (short)0, acc[tm][tn], false, false);

        if (kk + 1 < NK) store_stage(cur ^ 1);                // fill ping-pong buffer
        __syncthreads();
    }

    // ---- epilogue: write h and JumpReLU acts ----
#pragma unroll
    for (int tn = 0; tn < 2; ++tn) {
        const int gn  = f0 + waveN * 32 + tn * 16 + r;
        const float thr = __expf(tparam[l * kF + gn]);
#pragma unroll
        for (int tm = 0; tm < 4; ++tm) {
            const int mbase = t0 + waveM * 64 + tm * 16 + 8 * hi;
#pragma unroll
            for (int v = 0; v < 8; ++v) {
                const float hv = acc[tm][tn][v];
                const size_t off = (size_t)(mbase + v) * (kL * kF) + (size_t)l * kF + gn;
                h[off]    = hv;
                acts[off] = (hv > thr) ? hv : 0.0f;
            }
        }
    }
}

// ---------------------------------------------------------------------------
// Decoder: x_hat[t0:t0+128, j, d0:d0+128] = acts[t, 0:(j+1)*F] @ Wj
// where Wj row (i*F + f) = w_dec[i, j, f, :].  K = (j+1)*2048, contiguous in A.
// Same double-buffered pipeline.
// ---------------------------------------------------------------------------
__global__ void __launch_bounds__(256)
clt_decoder_wmma(const float* __restrict__ acts,
                 const float* __restrict__ w_dec,
                 float* __restrict__ x_hat)
{
    __shared__ __bf16 As[2][BM * LSTR];
    __shared__ __bf16 Bs[2][BN * LSTR];

    const int tid   = threadIdx.x;
    const int lane  = tid & 31;
    const int wave  = tid >> 5;
    const int r     = lane & 15;
    const int hi    = lane >> 4;
    const int waveM = wave & 1;
    const int waveN = wave >> 1;

    const int d0 = blockIdx.x * BN;
    const int t0 = blockIdx.y * BM;
    const int j  = (kL - 1) - (int)blockIdx.z;   // launch longest-K blocks first

    const float* Ag = acts + (size_t)t0 * (kL * kF);  // row stride kL*kF, K contiguous
    const int K = (j + 1) * kF;

    floatx8 zacc = {};
    floatx8 acc[4][2];
#pragma unroll
    for (int a = 0; a < 4; ++a)
#pragma unroll
        for (int b = 0; b < 2; ++b) acc[a][b] = zacc;

    const int a_row = tid >> 3;
    const int a_col = (tid & 7) << 2;
    const int nb = (tid & 63) << 1;
    const int kb = (tid >> 6) << 1;

    float4 va[4];
    float2 vblo[4], vbhi[4];

    auto load_stage = [&](int k0) {
        const int i = k0 >> 11;               // source layer (2048 % 32 == 0)
        const int f = k0 & (kF - 1);
        const float* Bg = w_dec + ((size_t)(i * kL + j) * kF + f) * kD + d0;  // row stride kD
#pragma unroll
        for (int p = 0; p < 4; ++p)
            va[p] = *(const float4*)(Ag + (size_t)(a_row + p * 32) * (kL * kF) + (k0 + a_col));
#pragma unroll
        for (int p = 0; p < 4; ++p) {
            const int k = kb + p * 8;
            vblo[p] = *(const float2*)(Bg + (size_t)(k)     * kD + nb);
            vbhi[p] = *(const float2*)(Bg + (size_t)(k + 1) * kD + nb);
        }
    };
    auto store_stage = [&](int buf) {
#pragma unroll
        for (int p = 0; p < 4; ++p) {
            __bf16* d = &As[buf][(a_row + p * 32) * LSTR + a_col];
            d[0] = (__bf16)va[p].x; d[1] = (__bf16)va[p].y;
            d[2] = (__bf16)va[p].z; d[3] = (__bf16)va[p].w;
        }
#pragma unroll
        for (int p = 0; p < 4; ++p) {
            const int k = kb + p * 8;
            bf16x2 c0, c1;
            c0[0] = (__bf16)vblo[p].x; c0[1] = (__bf16)vbhi[p].x;
            c1[0] = (__bf16)vblo[p].y; c1[1] = (__bf16)vbhi[p].y;
            *(bf16x2*)&Bs[buf][(nb + 0) * LSTR + k] = c0;
            *(bf16x2*)&Bs[buf][(nb + 1) * LSTR + k] = c1;
        }
    };

    load_stage(0);
    store_stage(0);
    __syncthreads();

    const int NK = K / BK;
    for (int kk = 0; kk < NK; ++kk) {
        const int cur = kk & 1;
        if (kk + 1 < NK) load_stage((kk + 1) * BK);
        if (kk + 2 < NK) {
            const int kp = (kk + 2) * BK;
            const int ip = kp >> 11;
            const int fp = kp & (kF - 1);
            __builtin_prefetch(Ag + (size_t)a_row * (kL * kF) + (kp + a_col), 0, 1);
            __builtin_prefetch(w_dec + ((size_t)(ip * kL + j) * kF + fp) * kD + d0 +
                                   (size_t)kb * kD + nb, 0, 1);
        }

        const __bf16* Ab = As[cur];
        const __bf16* Bb = Bs[cur];
        bf16x16 af[4];
#pragma unroll
        for (int tm = 0; tm < 4; ++tm) {
            const int m = waveM * 64 + tm * 16 + r;
            af[tm] = make_frag(&Ab[m * LSTR + hi * 8], &Ab[m * LSTR + 16 + hi * 8]);
        }
        bf16x16 bfr[2];
#pragma unroll
        for (int tn = 0; tn < 2; ++tn) {
            const int c = waveN * 32 + tn * 16 + r;
            bfr[tn] = make_frag(&Bb[c * LSTR + hi * 8], &Bb[c * LSTR + 16 + hi * 8]);
        }
#pragma unroll
        for (int tm = 0; tm < 4; ++tm)
#pragma unroll
            for (int tn = 0; tn < 2; ++tn)
                acc[tm][tn] = __builtin_amdgcn_wmma_f32_16x16x32_bf16(
                    false, af[tm], false, bfr[tn], (short)0, acc[tm][tn], false, false);

        if (kk + 1 < NK) store_stage(cur ^ 1);
        __syncthreads();
    }

#pragma unroll
    for (int tn = 0; tn < 2; ++tn) {
        const int gd = d0 + waveN * 32 + tn * 16 + r;
#pragma unroll
        for (int tm = 0; tm < 4; ++tm) {
            const int mbase = t0 + waveM * 64 + tm * 16 + 8 * hi;
#pragma unroll
            for (int v = 0; v < 8; ++v) {
                x_hat[(size_t)(mbase + v) * (kL * kD) + (size_t)j * kD + gd] = acc[tm][tn][v];
            }
        }
    }
}

// ---------------------------------------------------------------------------
// Launch: inputs = {x, w_e, w_dec, t, mask}; outputs = h | acts | x_hat flat.
// The triangular mask is implicit in the decoder's K-loop bound.
// ---------------------------------------------------------------------------
extern "C" void kernel_launch(void* const* d_in, const int* in_sizes, int n_in,
                              void* d_out, int out_size, void* d_ws, size_t ws_size,
                              hipStream_t stream) {
    const float* x      = (const float*)d_in[0];
    const float* w_e    = (const float*)d_in[1];
    const float* w_dec  = (const float*)d_in[2];
    const float* tparam = (const float*)d_in[3];

    float* h     = (float*)d_out;
    float* acts  = h    + (size_t)kT * kL * kF;
    float* x_hat = acts + (size_t)kT * kL * kF;

    dim3 block(256, 1, 1);
    dim3 genc(kF / BN, kT / BM, kL);   // 16 x 32 x 8
    clt_encoder_wmma<<<genc, block, 0, stream>>>(x, w_e, tparam, h, acts);

    dim3 gdec(kD / BN, kT / BM, kL);   // 6 x 32 x 8
    clt_decoder_wmma<<<gdec, block, 0, stream>>>(acts, w_dec, x_hat);
}
